// MultiHeadAttention_34746285425441
// MI455X (gfx1250) — matrix operand order repaired and checked by reading, not compiled, not run on hardware
//
#include <hip/hip_runtime.h>

#define BDIM 2
#define SEQ  2048
#define DM   1024
#define NH   16
#define HD   64

typedef __attribute__((ext_vector_type(16))) __bf16 v16bf;
typedef __attribute__((ext_vector_type(8)))  __bf16 v8bf;
typedef __attribute__((ext_vector_type(8)))  float  v8f;
typedef __attribute__((ext_vector_type(4)))  float  v4f;

static __device__ inline v8f wmma_bf16(v16bf a, v16bf b, v8f c) {
    // D = A(16x32 bf16) x B(32x16 bf16) + C(16x16 f32)
    return __builtin_amdgcn_wmma_f32_16x16x32_bf16(
        /*neg_a=*/false, a, /*neg_b=*/false, b,
        /*c_mod=*/(short)0, c, /*reuse_a=*/false, /*reuse_b=*/false);
}

// A-style fragment: two contiguous 8-element runs separated by 16 elements
static __device__ inline v16bf load_fragA(const __bf16* p) {
    v8bf lo = *(const v8bf*)p;
    v8bf hi = *(const v8bf*)(p + 16);
    v16bf f;
#pragma unroll
    for (int e = 0; e < 8; ++e) { f[e] = lo[e]; f[e + 8] = hi[e]; }
    return f;
}

// B-style fragment: two contiguous 8-element runs separated by 16 elements
static __device__ inline v16bf load_fragB(const __bf16* p) {
    v8bf lo = *(const v8bf*)p;
    v8bf hi = *(const v8bf*)(p + 16);
    v16bf f;
#pragma unroll
    for (int e = 0; e < 8; ++e) { f[e] = lo[e]; f[e + 8] = hi[e]; }
    return f;
}

// ---------------------------------------------------------------------------
// One-time fp32 -> bf16 conversion (8 elements/thread, b128 in / b128 out).
// ---------------------------------------------------------------------------
__global__ __launch_bounds__(256) void cvt_bf16(const float* __restrict__ src,
                                                __bf16* __restrict__ dst) {
    const size_t i = ((size_t)blockIdx.x * 256 + threadIdx.x) * 8;
    v4f f0 = *(const v4f*)(src + i);
    v4f f1 = *(const v4f*)(src + i + 4);
    v8bf o;
#pragma unroll
    for (int e = 0; e < 4; ++e) {
        o[e]     = static_cast<__bf16>(f0[e]);
        o[e + 4] = static_cast<__bf16>(f1[e]);
    }
    *(v8bf*)(dst + i) = o;
}

// ---------------------------------------------------------------------------
// Generic bf16 GEMM: C[m][n] = sum_k X[m][k] * W[n][k]  (X: 4096xDM, W: DMxDM)
// MODE 0: Q  -> bf16 [B][H][S][hd], pre-scaled by 1/sqrt(hd)
// MODE 1: K  -> bf16 [B][H][S][hd]
// MODE 2: V  -> bf16 [B][H][hd][S]   (transposed for P@V B-fragments)
// MODE 3: attn-out @ wo^T -> fp32 d_out [B][S][D]
// One wave computes a 16x64 strip. Double-buffered K-loop (step 64 = 2 WMMA
// stages): loads for stage n+1 are issued before stage n's WMMAs consume,
// hiding global latency behind the matrix pipe.
// ---------------------------------------------------------------------------
template<int MODE>
__global__ __launch_bounds__(128) void proj_wmma(const __bf16* __restrict__ X,
                                                 const __bf16* __restrict__ W,
                                                 void* __restrict__ outv) {
    const int lane = threadIdx.x & 31;
    const int ll   = lane & 15;
    const int kh   = lane >> 4;
    const int wid  = blockIdx.x * 4 + (threadIdx.x >> 5);
    const int mtile = wid >> 4;          // 0..255  (rows of 16)
    const int n0    = (wid & 15) * 64;   // 0..960  (strip of 64 cols)

    v8f acc[4];
#pragma unroll
    for (int j = 0; j < 4; ++j)
#pragma unroll
        for (int r = 0; r < 8; ++r) acc[j][r] = 0.0f;

    const int am = mtile * 16 + ll;
    const __bf16* Abase = X + (size_t)am * DM + kh * 8;
    const __bf16* Wrow[4];
#pragma unroll
    for (int j = 0; j < 4; ++j)
        Wrow[j] = W + (size_t)(n0 + j * 16 + ll) * DM + kh * 8;

    // prologue: stage 0
    v16bf a0 = load_fragA(Abase);
    v16bf b0[4];
#pragma unroll
    for (int j = 0; j < 4; ++j) b0[j] = load_fragB(Wrow[j]);

    for (int k0 = 0; k0 < DM; k0 += 64) {
        // issue stage-1 loads before consuming stage 0
        v16bf a1 = load_fragA(Abase + k0 + 32);
        v16bf b1[4];
#pragma unroll
        for (int j = 0; j < 4; ++j) b1[j] = load_fragB(Wrow[j] + k0 + 32);
#pragma unroll
        for (int j = 0; j < 4; ++j) acc[j] = wmma_bf16(a0, b0[j], acc[j]);

        // issue next stage-0 loads (wraps to 0 on the final trip: harmless)
        const int kn = (k0 + 64) & (DM - 1);
        a0 = load_fragA(Abase + kn);
#pragma unroll
        for (int j = 0; j < 4; ++j) b0[j] = load_fragB(Wrow[j] + kn);
#pragma unroll
        for (int j = 0; j < 4; ++j) acc[j] = wmma_bf16(a1, b1[j], acc[j]);
    }

#pragma unroll
    for (int j = 0; j < 4; ++j) {
#pragma unroll
        for (int r = 0; r < 8; ++r) {
            float val = acc[j][r];
            if constexpr (MODE == 0) val *= 0.125f;   // fold 1/sqrt(hd) into Q
            const int gm = mtile * 16 + r + kh * 8;   // row in [0, B*S)
            const int c  = n0 + j * 16 + ll;          // col in [0, D)
            if constexpr (MODE == 3) {
                ((float*)outv)[(size_t)gm * DM + c] = val;
            } else {
                const int b = gm >> 11, s = gm & (SEQ - 1);
                const int h = c >> 6,  d = c & (HD - 1);
                __bf16* O = (__bf16*)outv;
                if constexpr (MODE == 2)
                    O[(((size_t)(b * NH + h)) * HD + d) * SEQ + s] = static_cast<__bf16>(val);
                else
                    O[(((size_t)(b * NH + h)) * SEQ + s) * HD + d] = static_cast<__bf16>(val);
            }
        }
    }
}

// ---------------------------------------------------------------------------
// Causal flash attention. One wave owns one (b,h, 16-query) tile and streams
// key blocks of 32. All K and V fragments for a block are loaded in one
// clause at iteration start, so the softmax VALU section overlaps the V
// loads' latency. Scores: 2 tiles x 2 chained WMMA (K-depth 64). Online
// softmax in C-layout via 16-lane shfl_xor butterflies; P transposed to
// A-layout through per-wave LDS (no block barrier -> no deadlock on unequal
// causal trip counts). Q is pre-scaled by 1/sqrt(hd).
// ---------------------------------------------------------------------------
__global__ __launch_bounds__(128) void attn_flash(const __bf16* __restrict__ Q,
                                                  const __bf16* __restrict__ K,
                                                  const __bf16* __restrict__ VT,
                                                  __bf16* __restrict__ AO) {
    __shared__ __align__(16) __bf16 ldsP[4][16][32];

    const int lane = threadIdx.x & 31;
    const int ll   = lane & 15;
    const int kh   = lane >> 4;
    const int w    = threadIdx.x >> 5;
    const int wid  = blockIdx.x * 4 + w;
    const int bh   = wid >> 7;        // 0..31
    const int qt   = wid & 127;       // 0..127
    const int b = bh >> 4, h = bh & 15;

    const __bf16* Qb = Q  + (size_t)bh * SEQ * HD;
    const __bf16* Kb = K  + (size_t)bh * SEQ * HD;
    const __bf16* Vb = VT + (size_t)bh * HD * SEQ;

    v16bf aq[2];
#pragma unroll
    for (int c = 0; c < 2; ++c)
        aq[c] = load_fragA(Qb + (size_t)(qt * 16 + ll) * HD + c * 32 + kh * 8);

    v8f o[4];
    float mrow[8], lrow[8];
#pragma unroll
    for (int j = 0; j < 4; ++j)
#pragma unroll
        for (int r = 0; r < 8; ++r) o[j][r] = 0.0f;
#pragma unroll
    for (int r = 0; r < 8; ++r) { mrow[r] = -3.0e38f; lrow[r] = 0.0f; }

    const int qhi = qt * 16 + 15;
    for (int kb = 0; kb <= qhi; kb += 32) {
        if (kb + 32 <= qhi) {
            __builtin_prefetch(Kb + (size_t)(kb + 32) * HD, 0, 1);
            __builtin_prefetch(Vb + (size_t)(kb + 32), 0, 1);
        }
        // ---- load ALL fragments for this key block up front (one clause) ----
        v16bf bk[2][2];
#pragma unroll
        for (int t = 0; t < 2; ++t)
#pragma unroll
            for (int c = 0; c < 2; ++c)
                bk[t][c] = load_fragB(Kb + (size_t)(kb + t * 16 + ll) * HD + c * 32 + kh * 8);
        v16bf bv[4];
#pragma unroll
        for (int j = 0; j < 4; ++j)
            bv[j] = load_fragB(Vb + (size_t)(j * 16 + ll) * SEQ + kb + kh * 8);

        // ---- S = Q @ K^T : two 16x16 score tiles, K-depth 64 ----
        v8f sc[2];
#pragma unroll
        for (int t = 0; t < 2; ++t) {
#pragma unroll
            for (int r = 0; r < 8; ++r) sc[t][r] = 0.0f;
            sc[t] = wmma_bf16(aq[0], bk[t][0], sc[t]);
            sc[t] = wmma_bf16(aq[1], bk[t][1], sc[t]);
        }
        // causal mask (data-level select, EXEC stays all-ones for WMMA)
        float rmax[8], rsum[8];
#pragma unroll
        for (int r = 0; r < 8; ++r) {
            const int qi = qt * 16 + r + kh * 8;
#pragma unroll
            for (int t = 0; t < 2; ++t) {
                const int ki = kb + t * 16 + ll;
                sc[t][r] = (ki > qi) ? -1.0e30f : sc[t][r];
            }
            rmax[r] = fmaxf(sc[0][r], sc[1][r]);
        }
#pragma unroll
        for (int off = 8; off >= 1; off >>= 1)
#pragma unroll
            for (int r = 0; r < 8; ++r)
                rmax[r] = fmaxf(rmax[r], __shfl_xor(rmax[r], off, 32));

        float alpha[8];
#pragma unroll
        for (int r = 0; r < 8; ++r) {
            const float mn = fmaxf(mrow[r], rmax[r]);
            alpha[r] = __expf(mrow[r] - mn);
            mrow[r]  = mn;
            const float p0 = __expf(sc[0][r] - mn);
            const float p1 = __expf(sc[1][r] - mn);
            sc[0][r] = p0; sc[1][r] = p1;
            rsum[r] = p0 + p1;
        }
#pragma unroll
        for (int off = 8; off >= 1; off >>= 1)
#pragma unroll
            for (int r = 0; r < 8; ++r)
                rsum[r] += __shfl_xor(rsum[r], off, 32);
#pragma unroll
        for (int r = 0; r < 8; ++r) lrow[r] = lrow[r] * alpha[r] + rsum[r];

        // P: C-layout -> A-layout via per-wave LDS tile (wave-private sync)
#pragma unroll
        for (int r = 0; r < 8; ++r) {
            const int row = r + kh * 8;
            ldsP[w][row][ll]      = static_cast<__bf16>(sc[0][r]);
            ldsP[w][row][ll + 16] = static_cast<__bf16>(sc[1][r]);
        }
        asm volatile("s_wait_dscnt 0" ::: "memory");
        v16bf pa;
        {
            const __bf16* pr = &ldsP[w][ll][kh * 8];
            v8bf lo = *(const v8bf*)pr;
            v8bf hi = *(const v8bf*)(pr + 16);
#pragma unroll
            for (int e = 0; e < 8; ++e) { pa[e] = lo[e]; pa[e + 8] = hi[e]; }
        }
        // O = O*alpha + P @ V  (V fragments already resident in registers)
#pragma unroll
        for (int j = 0; j < 4; ++j) {
#pragma unroll
            for (int r = 0; r < 8; ++r) o[j][r] *= alpha[r];
            o[j] = wmma_bf16(pa, bv[j], o[j]);
        }
    }

    // normalize and emit attention output as bf16 [B][S][D]
#pragma unroll
    for (int r = 0; r < 8; ++r) {
        const float inv = 1.0f / lrow[r];
        const int sq = qt * 16 + r + kh * 8;
#pragma unroll
        for (int j = 0; j < 4; ++j) {
            const int cD = h * HD + j * 16 + ll;
            AO[((size_t)(b * SEQ + sq)) * DM + cD] = static_cast<__bf16>(o[j][r] * inv);
        }
    }
}

extern "C" void kernel_launch(void* const* d_in, const int* in_sizes, int n_in,
                              void* d_out, int out_size, void* d_ws, size_t ws_size,
                              hipStream_t stream) {
    (void)in_sizes; (void)n_in; (void)out_size; (void)ws_size;
    const float* x  = (const float*)d_in[0];
    const float* wq = (const float*)d_in[1];
    const float* wk = (const float*)d_in[2];
    const float* wv = (const float*)d_in[3];
    const float* wo = (const float*)d_in[4];
    float* out = (float*)d_out;

    const size_t x_elems = (size_t)BDIM * SEQ * DM;      // 4,194,304
    const size_t w_elems = (size_t)DM * DM;              // 1,048,576
    __bf16* xbf  = (__bf16*)d_ws;
    __bf16* wqbf = xbf  + x_elems;
    __bf16* wkbf = wqbf + w_elems;
    __bf16* wvbf = wkbf + w_elems;
    __bf16* wobf = wvbf + w_elems;
    __bf16* qws  = wobf + w_elems;
    __bf16* kws  = qws  + x_elems;
    __bf16* vtws = kws  + x_elems;
    __bf16* aws  = vtws + x_elems;   // total ~48 MB of workspace

    dim3 blk(128);  // 4 waves (wave32)
    cvt_bf16<<<2048, 256, 0, stream>>>(x,  xbf);
    cvt_bf16<<< 512, 256, 0, stream>>>(wq, wqbf);
    cvt_bf16<<< 512, 256, 0, stream>>>(wk, wkbf);
    cvt_bf16<<< 512, 256, 0, stream>>>(wv, wvbf);
    cvt_bf16<<< 512, 256, 0, stream>>>(wo, wobf);

    proj_wmma<0><<<1024, blk, 0, stream>>>(xbf, wqbf, qws);
    proj_wmma<1><<<1024, blk, 0, stream>>>(xbf, wkbf, kws);
    proj_wmma<2><<<1024, blk, 0, stream>>>(xbf, wvbf, vtws);
    attn_flash <<<1024, blk, 0, stream>>>(qws, kws, vtws, aws);
    proj_wmma<3><<<1024, blk, 0, stream>>>(aws, wobf, out);
}
